// LSTM_16381005267368
// MI455X (gfx1250) — matrix operand order repaired and checked by reading, not compiled, not run on hardware
//
#include <hip/hip_runtime.h>
#include <hip/hip_bf16.h>

typedef __bf16 bf16;
typedef __attribute__((ext_vector_type(16))) __bf16 v16bf;
typedef __attribute__((ext_vector_type(8)))  float  v8f;

#define T_STEPS 2048
#define NMT 4                      // B/16 row tiles
#define NKT 16                     // 512/32 K tiles
#define NHT 32                     // H/16 column tiles
#define FRAG 512                   // bf16 elems per 16x32 / 32x16 fragment (32 lanes * 16)
#define APER_T (NMT*NKT*FRAG)      // 32768 bf16 per timestep of packed activations
#define WL_STRIDE (NHT*NKT*4*FRAG) // 1048576 bf16 per-layer packed weights
#define XG_FRAG 256                // f32 elems per 16x16 D fragment (32 lanes * 8)
#define TT 32                      // timesteps per gemm_xg block

// ---------- WMMA helper (CDNA5: D = A(16x32,bf16) * B(32x16,bf16) + C(f32)) ----------
__device__ __forceinline__ v8f wmma_bf16(v16bf a, v16bf b, v8f c) {
  return __builtin_amdgcn_wmma_f32_16x16x32_bf16(false, a, false, b, (short)0, c, false, false);
}

__device__ __forceinline__ float fsigm(float x) {
  return __builtin_amdgcn_rcpf(1.0f + __expf(-x));
}
__device__ __forceinline__ float ftanh_(float x) {
  return 2.0f * __builtin_amdgcn_rcpf(1.0f + __expf(-2.0f * x)) - 1.0f;
}

// ---------- counter init (workspace is poisoned each run) ----------
__global__ void init_ctr(unsigned* ctr) {
  if (threadIdx.x < 64) ctr[threadIdx.x] = 0u;
}

// ---------- pack Wx/Wh into B-fragment layout ----------
// B 32x16 bf16 layout: lane = n + 16*((k>>3)&1), slot = (k&7) + 8*(k>>4)
// buffer layout: [l][ht][kk][g][lane*16+slot]
__global__ void pack_w(const float* __restrict__ Wx, const float* __restrict__ Wh,
                       bf16* __restrict__ wxf, bf16* __restrict__ whf) {
  unsigned idx  = blockIdx.x * 256u + threadIdx.x;
  unsigned slot = idx & 15u;
  unsigned lane = (idx >> 4) & 31u;
  unsigned g    = (idx >> 9) & 3u;
  unsigned kk   = (idx >> 11) & 15u;
  unsigned ht   = (idx >> 15) & 31u;
  unsigned l    = (idx >> 20) & 1u;
  unsigned ncol = lane & 15u, khi = lane >> 4;
  unsigned kloc = (slot & 7u) + 8u * khi + 16u * (slot >> 3);
  unsigned k    = kk * 32u + kloc;
  unsigned hcol = ht * 16u + ncol;
  size_t src = ((size_t)(l * 4u + g) * 512u + hcol) * 512u + k; // [L,4,H,{I|H}]
  wxf[idx] = (bf16)Wx[src];
  whf[idx] = (bf16)Wh[src];
}

// ---------- pack activations [T,B,512] f32 -> A-fragment bf16 layout ----------
// A 16x32 bf16: lane = m + 16*((k>>3)&1), slot = (k&7) + 8*(k>>4)
// buffer layout: [t][mt][kt][lane*16+slot]
__global__ void pack_a(const float* __restrict__ src, bf16* __restrict__ dst) {
  size_t   idx  = (size_t)blockIdx.x * 256u + threadIdx.x;
  unsigned slot = (unsigned)idx & 15u;
  unsigned lane = (unsigned)(idx >> 4) & 31u;
  unsigned kt   = (unsigned)(idx >> 9) & 15u;
  unsigned mt   = (unsigned)(idx >> 13) & 3u;
  unsigned t    = (unsigned)(idx >> 15);
  unsigned m  = lane & 15u, hi = lane >> 4;
  unsigned k  = (slot & 7u) | (hi << 3) | ((slot >> 3) << 4);
  unsigned b  = mt * 16u + m;
  unsigned i  = kt * 32u + k;
  dst[idx] = (bf16)src[((size_t)t * 64u + b) * 512u + i];
}

// ---------- input-projection GEMM: xg[t] = act[t] @ Wx^T (fp32 D-frag layout) ----------
// grid = (T/TT, NHT). Block parks its ht's Wx B-fragments (64 KB) in LDS,
// streams pipelined A fragments from global. LDS offset is laundered through
// inline asm each timestep so the compiler cannot hoist the B reads into
// registers (which previously caused scratch spills).
__global__ void __launch_bounds__(256) gemm_xg(const bf16* __restrict__ afrag,
                                               const bf16* __restrict__ wxf_l,
                                               float* __restrict__ xg) {
  __shared__ bf16 wlds[NKT * 4 * FRAG];      // 64 KB weights-stationary
  const unsigned ttile = blockIdx.x;
  const unsigned ht    = blockIdx.y;
  const unsigned tid   = threadIdx.x;
  {
    const uint4* s = (const uint4*)(wxf_l + (size_t)ht * NKT * 4u * FRAG);
    uint4*       d = (uint4*)wlds;
    for (unsigned i = tid; i < (NKT * 4 * FRAG) / 8; i += 256u) d[i] = s[i];
  }
  __syncthreads();

  const unsigned wave = tid >> 5, lane = tid & 31u;
  const unsigned mt = wave & 3u, tsub = wave >> 2;

  for (unsigned i = 0; i < TT / 2; ++i) {
    const unsigned t = ttile * TT + i * 2u + tsub;
    const bf16* ap = afrag + (size_t)t * APER_T + (size_t)mt * NKT * FRAG + lane * 16u;
    // Opaque per-iteration LDS offset: keeps B reads as in-loop ds_loads.
    unsigned lbase = lane * 16u;
    asm volatile("" : "+v"(lbase));
    v8f acc[4] = {};
    v16bf a0 = *(const v16bf*)ap;           // software-pipelined A stream
#pragma unroll
    for (int kk = 0; kk < 16; ++kk) {
      v16bf an = a0;
      if (kk < 15) an = *(const v16bf*)(ap + (size_t)(kk + 1) * FRAG);
#pragma unroll
      for (int g = 0; g < 4; ++g) {
        v16bf b = *(const v16bf*)(&wlds[(kk * 4 + g) * FRAG + lbase]);
        acc[g] = wmma_bf16(a0, b, acc[g]);
      }
      a0 = an;
    }
    float* xo = xg + ((size_t)(t * 4u + mt) * 32u + ht) * 4u * XG_FRAG + lane * 8u;
#pragma unroll
    for (int g = 0; g < 4; ++g) *(v8f*)(xo + g * XG_FRAG) = acc[g];
  }
}

// ---------- persistent recurrent kernel: one grid barrier per timestep ----------
// grid = 32 blocks (ht), block = 128 threads = 4 waves (mt). Wh slice lives in LDS.
__global__ void __launch_bounds__(128) lstm_recur(
    const bf16* __restrict__ hread, unsigned hmod,
    bf16* __restrict__ hwrite,
    const bf16* __restrict__ whf_l,
    const float* __restrict__ xg,
    const float* __restrict__ bx_l,
    float* __restrict__ out,          // nullptr for hidden layers
    unsigned* ctr) {
  __shared__ bf16 wlds[NKT * 4 * FRAG]; // 64 KB weights-stationary
  const unsigned ht  = blockIdx.x;
  const unsigned tid = threadIdx.x;
  {
    const uint4* s = (const uint4*)(whf_l + (size_t)ht * NKT * 4u * FRAG);
    uint4*       d = (uint4*)wlds;
    for (unsigned i = tid; i < (NKT * 4 * FRAG) / 8; i += 128u) d[i] = s[i];
  }
  __syncthreads();

  const unsigned wave = tid >> 5, lane = tid & 31u, mt = wave;
  const unsigned ncol = lane & 15u, mhi = lane >> 4;
  const unsigned hcol = ht * 16u + ncol;
  const float bi  = bx_l[hcol];
  const float bff = bx_l[512u + hcol];
  const float bo  = bx_l[1024u + hcol];
  const float bc  = bx_l[1536u + hcol];
  // h-fragment write address components (A-layout inverse; constant per lane)
  const unsigned kt_w   = hcol >> 5, kloc = hcol & 31u;
  const unsigned lanehi = (kloc >> 3) & 1u;
  const unsigned slot_w = (kloc & 7u) + 8u * (kloc >> 4);

  v8f c = {};
  unsigned target = 0;
  for (unsigned t = 0; t < T_STEPS; ++t) {
    // xg does not depend on the barrier -> issue these loads early (hide under spin)
    v8f acc[4];
    const float* xgp = xg + ((size_t)(t * 4u + mt) * 32u + ht) * 4u * XG_FRAG + lane * 8u;
#pragma unroll
    for (int g = 0; g < 4; ++g) acc[g] = *(const v8f*)(xgp + g * XG_FRAG);

    if (t > 0) {
      // ---- grid barrier: h(t-1) fragments from all blocks are visible after this ----
      target += gridDim.x;
      __syncthreads();
      if (tid == 0) {
        __threadfence();
        __hip_atomic_fetch_add(ctr, 1u, __ATOMIC_RELEASE, __HIP_MEMORY_SCOPE_AGENT);
        while (__hip_atomic_load(ctr, __ATOMIC_ACQUIRE, __HIP_MEMORY_SCOPE_AGENT) < target)
          __builtin_amdgcn_s_sleep(1);
      }
      __syncthreads();
      // ---- recurrent GEMM: 16 K-steps x 4 gates of v_wmma_f32_16x16x32_bf16 ----
      const bf16* hp = hread + (size_t)((t - 1) % hmod) * APER_T +
                       (size_t)mt * NKT * FRAG + lane * 16u;
      // Opaque per-iteration LDS offset: keeps B reads as in-loop ds_loads.
      unsigned lbase = lane * 16u;
      asm volatile("" : "+v"(lbase));
      v16bf a0 = *(const v16bf*)hp;       // software-pipelined A stream
#pragma unroll
      for (int kk = 0; kk < 16; ++kk) {
        v16bf an = a0;
        if (kk < 15) an = *(const v16bf*)(hp + (size_t)(kk + 1) * FRAG);
#pragma unroll
        for (int g = 0; g < 4; ++g) {
          v16bf b = *(const v16bf*)(&wlds[(kk * 4 + g) * FRAG + lbase]);
          acc[g] = wmma_bf16(a0, b, acc[g]);
        }
        a0 = an;
      }
    }

    // ---- elementwise gates on D-fragments (gate order i,f,o,c) ----
    float hvals[8];
#pragma unroll
    for (int r = 0; r < 8; ++r) {
      float iv = fsigm(acc[0][r] + bi);
      float ov = fsigm(acc[2][r] + bo);
      float cc = ftanh_(acc[3][r] + bc);
      float cn;
      if (t > 0) { float fv = fsigm(acc[1][r] + bff); cn = fv * c[r] + iv * cc; }
      else        { cn = iv * cc; }                   // t==0: no forget gate
      c[r]     = cn;
      hvals[r] = ov * ftanh_(cn);
    }

    // ---- write h as bf16 A-fragments for step t+1 / next layer ----
    bf16* hw = hwrite + (size_t)(t % hmod) * APER_T +
               (size_t)mt * NKT * FRAG + (size_t)kt_w * FRAG;
#pragma unroll
    for (int r = 0; r < 8; ++r) {
      unsigned m = (unsigned)r + 8u * mhi;
      hw[(m + 16u * lanehi) * 16u + slot_w] = (bf16)hvals[r];
    }
    if (out) {
#pragma unroll
      for (int r = 0; r < 8; ++r) {
        unsigned b = mt * 16u + (unsigned)r + 8u * mhi;
        out[((size_t)t * 64u + b) * 512u + hcol] = hvals[r];
      }
    }
  }
}

extern "C" void kernel_launch(void* const* d_in, const int* in_sizes, int n_in,
                              void* d_out, int out_size, void* d_ws, size_t ws_size,
                              hipStream_t stream) {
  (void)in_sizes; (void)n_in; (void)out_size; (void)ws_size;
  const float* x  = (const float*)d_in[0];   // [T,B,I]
  const float* Wx = (const float*)d_in[1];   // [L,4,H,I]
  const float* bx = (const float*)d_in[2];   // [L,4,H]
  const float* Wh = (const float*)d_in[3];   // [L,4,H,H]
  float* out = (float*)d_out;                // [T,B,H]

  // workspace carve-up (all chunks 256B-aligned by construction)
  char* ws = (char*)d_ws;
  unsigned* ctr = (unsigned*)ws;                               // 256 B: barrier counters
  bf16* wxf   = (bf16*)(ws + 256);                             // 2 * WL_STRIDE
  bf16* whf   = wxf + (size_t)2 * WL_STRIDE;                   // 2 * WL_STRIDE
  bf16* xfrag = whf + (size_t)2 * WL_STRIDE;                   // T * APER_T
  bf16* hfrag = xfrag + (size_t)T_STEPS * APER_T;              // T * APER_T
  bf16* hbuf  = hfrag + (size_t)T_STEPS * APER_T;              // 2 * APER_T
  float* xg   = (float*)(hbuf + (size_t)2 * APER_T);           // T * XG_PER_T f32 (1 GiB)

  hipLaunchKernelGGL(init_ctr, dim3(1), dim3(64), 0, stream, ctr);
  hipLaunchKernelGGL(pack_w, dim3((2u * WL_STRIDE) / 256u), dim3(256), 0, stream,
                     Wx, Wh, wxf, whf);
  hipLaunchKernelGGL(pack_a, dim3((unsigned)(((size_t)T_STEPS * APER_T) / 256u)), dim3(256),
                     0, stream, x, xfrag);

  // ---- layer 1: parallel x-projection, then sequential recurrence ----
  hipLaunchKernelGGL(gemm_xg, dim3(T_STEPS / TT, NHT), dim3(256), 0, stream,
                     xfrag, wxf, xg);
  hipLaunchKernelGGL(lstm_recur, dim3(NHT), dim3(128), 0, stream,
                     (const bf16*)hfrag, (unsigned)T_STEPS, hfrag, whf, xg, bx,
                     (float*)nullptr, ctr);

  // ---- layer 2: projection from layer-1 h fragments, recurrence writes d_out ----
  hipLaunchKernelGGL(gemm_xg, dim3(T_STEPS / TT, NHT), dim3(256), 0, stream,
                     hfrag, wxf + (size_t)WL_STRIDE, xg);
  hipLaunchKernelGGL(lstm_recur, dim3(NHT), dim3(128), 0, stream,
                     (const bf16*)hbuf, 2u, hbuf, whf + (size_t)WL_STRIDE, xg,
                     bx + 4 * 512, out, ctr + 1);
}
